// CausalSelfAttention_1133871366572
// MI455X (gfx1250) — matrix operand order repaired and checked by reading, not compiled, not run on hardware
//
#include <hip/hip_runtime.h>
#include <hip/hip_bf16.h>
#include <stdint.h>

// ---------------------------------------------------------------------------
// CDNA5 (gfx1250) causal self-attention.
// Pipeline: cvt(x)->f16, transpose+cvt(weights)->f16  (one-time prep)
//           QKV GEMM (all-async LDS staging, double-buffered, WMMA f16)
//           flash attention (async K staging, WMMA QK^T and PV)
//           out GEMM (same all-async WMMA GEMM)
// ---------------------------------------------------------------------------

typedef __attribute__((ext_vector_type(16))) _Float16 v16h;
typedef __attribute__((ext_vector_type(4)))  _Float16 v4h;
typedef __attribute__((ext_vector_type(8)))  float    v8f;

#define WMMA_F16(a, b, c) \
  __builtin_amdgcn_wmma_f32_16x16x32_f16(false, (a), false, (b), (short)0, (c), false, false)

// A-matrix 16x32 f16 fragment (ISA 7.12.2): lane holds row m=lane&15,
// K contiguous: k = g*16 + i (g=lane>>4), matching load_frag_b.
__device__ __forceinline__ v16h load_frag_a(const _Float16* p0, int stride) {
  const int lane = threadIdx.x & 31;
  const int row  = lane & 15;
  const int g    = lane >> 4;
  const _Float16* p = p0 + row * stride + g * 16;
  v16h f;
#pragma unroll
  for (int i = 0; i < 16; ++i) f[i] = p[i];
  return f;
}

// B-matrix 32x16 f16 fragment: lane holds column n=lane&15,
// K contiguous: k = (lane>>4)*16 + i (this code's order, same as load_frag_a).
// Expects B stored column-major in LDS: p0[n*stride + k].
__device__ __forceinline__ v16h load_frag_b(const _Float16* p0, int stride) {
  const int lane = threadIdx.x & 31;
  const int col  = lane & 15;
  const int g    = lane >> 4;
  const _Float16* p = p0 + col * stride + g * 16;
  v16h f;
#pragma unroll
  for (int i = 0; i < 16; ++i) f[i] = p[i];
  return f;
}

// Async 16B copy global->LDS (GLOBAL_LOAD_ASYNC_TO_LDS_B128, ASYNCcnt).
// LDS offset = low 32 bits of the generic pointer (flat->LDS uses addr[31:0]).
__device__ __forceinline__ void async_b128(const _Float16* gp, _Float16* lp) {
  unsigned lds = (unsigned)(uintptr_t)lp;
  unsigned long long ga = (unsigned long long)(uintptr_t)gp;
  asm volatile("global_load_async_to_lds_b128 %0, %1, off"
               :: "v"(lds), "v"(ga) : "memory");
}
__device__ __forceinline__ void wait_async0() {
  asm volatile("s_wait_asynccnt 0" ::: "memory");
}

// ---------------------------------------------------------------------------
// Prep: elementwise f32 -> f16 (4 elements / thread, vector ld/st)
// ---------------------------------------------------------------------------
__global__ __launch_bounds__(256) void cvt_f16_kernel(const float* __restrict__ X,
                                                      _Float16* __restrict__ XH) {
  size_t i = ((size_t)blockIdx.x * 256 + threadIdx.x) * 4;
  float4 v = *(const float4*)(X + i);
  v4h h = {(_Float16)v.x, (_Float16)v.y, (_Float16)v.z, (_Float16)v.w};
  *(v4h*)(XH + i) = h;
}

// Prep: W[K][N] f32 -> WT[N][K] f16 via LDS tile transpose (64x64 tiles)
__global__ __launch_bounds__(256) void transpose_cvt_kernel(
    const float* __restrict__ W, _Float16* __restrict__ WT, int K, int N) {
  __shared__ float tile[64][65];
  const int k0 = blockIdx.y * 64, n0 = blockIdx.x * 64;
#pragma unroll
  for (int i = 0; i < 16; ++i) {
    int e = i * 256 + threadIdx.x;
    int r = e >> 6, c = e & 63;  // r: k, c: n (coalesced along n)
    tile[r][c] = W[(size_t)(k0 + r) * N + (n0 + c)];
  }
  __syncthreads();
#pragma unroll
  for (int i = 0; i < 16; ++i) {
    int e = i * 256 + threadIdx.x;
    int r = e >> 6, c = e & 63;  // r: n, c: k (coalesced along k)
    WT[(size_t)(n0 + r) * K + (k0 + c)] = (_Float16)tile[c][r];
  }
}

// ---------------------------------------------------------------------------
// GEMM: C[M,N] = A[M,K] * WT[N,K]^T + bias[N].  A, WT f16; staging is pure
// async DMA into double-buffered LDS; 8 waves x (32x32 tile), 8 WMMA / stage.
// ---------------------------------------------------------------------------
__device__ __forceinline__ void stage_tiles_async(
    const _Float16* A, const _Float16* BT,
    _Float16 As[][72], _Float16 Bs[][72],
    int rowBase, int colBase, int K, int k0, int tid) {
#pragma unroll
  for (int i = 0; i < 2; ++i) {  // A: 64x64 f16 = 512 x 16B chunks
    int e = i * 256 + tid;
    int m = e >> 3, c8 = (e & 7) * 8;
    async_b128(A + (size_t)(rowBase + m) * K + (k0 + c8), &As[m][c8]);
  }
#pragma unroll
  for (int i = 0; i < 4; ++i) {  // B: 128x64 f16 = 1024 x 16B chunks
    int e = i * 256 + tid;
    int n = e >> 3, c8 = (e & 7) * 8;
    async_b128(BT + (size_t)(colBase + n) * K + (k0 + c8), &Bs[n][c8]);
  }
}

template <typename CT>
__global__ __launch_bounds__(256) void gemm_wmma_kernel(
    const _Float16* __restrict__ A, const _Float16* __restrict__ BT,
    const float* __restrict__ bias, CT* __restrict__ C,
    int M, int N, int K) {
  constexpr int BM = 64, BN = 128, BK = 64, LDT = BK + 8;
  __shared__ __align__(16) _Float16 As[2][BM][LDT];
  __shared__ __align__(16) _Float16 Bs[2][BN][LDT];

  const int tid  = threadIdx.x;
  const int wave = tid >> 5;
  const int lane = tid & 31;
  const int wrow = wave >> 2;  // 0..1
  const int wcol = wave & 3;   // 0..3
  const int rowBase = blockIdx.y * BM;
  const int colBase = blockIdx.x * BN;

  v8f z = {};
  v8f acc[2][2];
#pragma unroll
  for (int i = 0; i < 2; ++i)
#pragma unroll
    for (int j = 0; j < 2; ++j) acc[i][j] = z;

  stage_tiles_async(A, BT, As[0], Bs[0], rowBase, colBase, K, 0, tid);
  wait_async0();
  __syncthreads();

  int buf = 0;
  for (int k0 = 0; k0 < K; k0 += BK) {
    if (k0 + BK < K)  // prefetch next tile into other buffer (overlaps WMMA)
      stage_tiles_async(A, BT, As[buf ^ 1], Bs[buf ^ 1],
                        rowBase, colBase, K, k0 + BK, tid);
#pragma unroll
    for (int kk = 0; kk < BK; kk += 32) {
      v16h a0 = load_frag_a(&As[buf][wrow * 32][kk], LDT);
      v16h a1 = load_frag_a(&As[buf][wrow * 32 + 16][kk], LDT);
      v16h b0 = load_frag_b(&Bs[buf][wcol * 32][kk], LDT);
      v16h b1 = load_frag_b(&Bs[buf][wcol * 32 + 16][kk], LDT);
      acc[0][0] = WMMA_F16(a0, b0, acc[0][0]);
      acc[0][1] = WMMA_F16(a0, b1, acc[0][1]);
      acc[1][0] = WMMA_F16(a1, b0, acc[1][0]);
      acc[1][1] = WMMA_F16(a1, b1, acc[1][1]);
    }
    wait_async0();
    __syncthreads();
    buf ^= 1;
  }

  // C/D layout: vgpr r -> row r + 8*(lane>>4); col = lane&15
  const int g  = lane >> 4;
  const int nn = lane & 15;
#pragma unroll
  for (int mi = 0; mi < 2; ++mi)
#pragma unroll
    for (int ni = 0; ni < 2; ++ni)
#pragma unroll
      for (int r = 0; r < 8; ++r) {
        int m   = rowBase + wrow * 32 + mi * 16 + r + 8 * g;
        int col = colBase + wcol * 32 + ni * 16 + nn;
        float v = acc[mi][ni][r] + bias[col];
        C[(size_t)m * N + col] = (CT)v;
      }
}

// ---------------------------------------------------------------------------
// Flash attention: grid (T/64, H, B), 128 threads (4 waves x 16 query rows).
// qkv layout: (B, T, 3*D) f16; head h: Q at col h*64, K at D+h*64, V at 2D+h*64.
// ---------------------------------------------------------------------------
__global__ __launch_bounds__(128) void flash_attn_kernel(
    const _Float16* __restrict__ qkv, _Float16* __restrict__ attn) {
  constexpr int T = 2048, D = 1024, HD = 64, LDT = HD + 8;
  const int b     = blockIdx.z;
  const int h     = blockIdx.y;
  const int qbase = blockIdx.x * 64;

  __shared__ __align__(16) _Float16 Qs[64][LDT];
  __shared__ __align__(16) _Float16 Ks[64][LDT];     // row-major: Ks[key][d]
  __shared__ __align__(16) _Float16 Vs[64][LDT];     // transposed: Vs[d][key]
  __shared__ __align__(16) _Float16 Ps[4][16][LDT];  // wave-private P tiles

  const int tid  = threadIdx.x;
  const int wave = tid >> 5;
  const int lane = tid & 31;
  const size_t rowStride = (size_t)3 * D;
  const _Float16* qp = qkv + ((size_t)b * T + qbase) * rowStride + h * HD;

  // Q block (64x64) via async DMA to LDS
#pragma unroll
  for (int p = 0; p < 4; ++p) {
    int r  = p * 16 + (tid >> 3);
    int c8 = (tid & 7) * 8;
    async_b128(qp + (size_t)r * rowStride + c8, &Qs[r][c8]);
  }
  wait_async0();
  __syncthreads();
  v16h aq0 = load_frag_a(&Qs[wave * 16][0], LDT);
  v16h aq1 = load_frag_a(&Qs[wave * 16][32], LDT);

  const float NEG_INF = -__builtin_inff();
  float mrun[8], lrun[8];
  v8f z = {};
  v8f o[4];
#pragma unroll
  for (int r = 0; r < 8; ++r) { mrun[r] = NEG_INF; lrun[r] = 0.0f; }
#pragma unroll
  for (int nt = 0; nt < 4; ++nt) o[nt] = z;

  const int g  = lane >> 4;
  const int nn = lane & 15;

  const int jmax = (qbase + 63) >> 6;  // causal: only key blocks <= query block
  for (int j = 0; j <= jmax; ++j) {
    __syncthreads();
    const _Float16* kp = qkv + ((size_t)b * T + j * 64) * rowStride + D + h * HD;
    const _Float16* vp = kp + D;
    // K block row-major via async DMA
#pragma unroll
    for (int p = 0; p < 4; ++p) {
      int r  = p * 16 + (tid >> 3);
      int c8 = (tid & 7) * 8;
      async_b128(kp + (size_t)r * rowStride + c8, &Ks[r][c8]);
    }
    // V block transposed into Vs[d][key] (scalar; scatter can't use DMA)
#pragma unroll
    for (int i = 0; i < 32; ++i) {
      int e = i * 128 + tid;
      int kk = e >> 6, d = e & 63;
      Vs[d][kk] = vp[(size_t)kk * rowStride + d];
    }
    wait_async0();
    __syncthreads();

    // S = Q K^T : B-matrix column n == row n of K -> load_frag_b on Ks rows
    v8f s[4];
#pragma unroll
    for (int nt = 0; nt < 4; ++nt) {
      v8f c = z;
      c = WMMA_F16(aq0, load_frag_b(&Ks[nt * 16][0], LDT), c);
      c = WMMA_F16(aq1, load_frag_b(&Ks[nt * 16][32], LDT), c);
      s[nt] = c;
    }

    // scale + causal mask
#pragma unroll
    for (int nt = 0; nt < 4; ++nt) {
      int kg = j * 64 + nt * 16 + nn;
#pragma unroll
      for (int r = 0; r < 8; ++r) {
        int qg = qbase + wave * 16 + r + 8 * g;
        float sv = s[nt][r] * 0.125f;  // 1/sqrt(64)
        s[nt][r] = (kg > qg) ? NEG_INF : sv;
      }
    }

    // online softmax; row r+8g lives in the 16-lane half 'g' -> xor masks 1..8
    float mnew[8], alpha[8];
#pragma unroll
    for (int r = 0; r < 8; ++r) {
      float mv = s[0][r];
#pragma unroll
      for (int nt = 1; nt < 4; ++nt) mv = fmaxf(mv, s[nt][r]);
#pragma unroll
      for (int mk = 1; mk < 16; mk <<= 1) mv = fmaxf(mv, __shfl_xor(mv, mk, 32));
      mnew[r]  = fmaxf(mrun[r], mv);
      alpha[r] = __expf(mrun[r] - mnew[r]);
      mrun[r]  = mnew[r];
    }
    float rsum[8];
#pragma unroll
    for (int r = 0; r < 8; ++r) rsum[r] = 0.0f;
#pragma unroll
    for (int nt = 0; nt < 4; ++nt)
#pragma unroll
      for (int r = 0; r < 8; ++r) {
        float pv = __expf(s[nt][r] - mnew[r]);
        s[nt][r] = pv;
        rsum[r] += pv;
      }
#pragma unroll
    for (int r = 0; r < 8; ++r) {
      float t = rsum[r];
#pragma unroll
      for (int mk = 1; mk < 16; mk <<= 1) t += __shfl_xor(t, mk, 32);
      lrun[r] = lrun[r] * alpha[r] + t;
    }
#pragma unroll
    for (int nt = 0; nt < 4; ++nt)
#pragma unroll
      for (int r = 0; r < 8; ++r) o[nt][r] *= alpha[r];

    // C-layout P -> LDS -> A-layout fragments (wave-private, in-order LDS)
#pragma unroll
    for (int nt = 0; nt < 4; ++nt)
#pragma unroll
      for (int r = 0; r < 8; ++r)
        Ps[wave][r + 8 * g][nt * 16 + nn] = (_Float16)s[nt][r];
    asm volatile("s_wait_dscnt 0" ::: "memory");
    v16h ap0 = load_frag_a(&Ps[wave][0][0], LDT);
    v16h ap1 = load_frag_a(&Ps[wave][0][32], LDT);

    // O += P V : B column n == hd index -> load_frag_b on transposed Vs
#pragma unroll
    for (int nt = 0; nt < 4; ++nt) {
      o[nt] = WMMA_F16(ap0, load_frag_b(&Vs[nt * 16][0], LDT), o[nt]);
      o[nt] = WMMA_F16(ap1, load_frag_b(&Vs[nt * 16][32], LDT), o[nt]);
    }
  }

  // epilogue: normalize and store f16
#pragma unroll
  for (int nt = 0; nt < 4; ++nt)
#pragma unroll
    for (int r = 0; r < 8; ++r) {
      int qg = qbase + wave * 16 + r + 8 * g;
      float val = o[nt][r] / lrun[r];
      attn[((size_t)b * T + qg) * (size_t)D + h * HD + nt * 16 + nn] = (_Float16)val;
    }
}

// ---------------------------------------------------------------------------
extern "C" void kernel_launch(void* const* d_in, const int* in_sizes, int n_in,
                              void* d_out, int out_size, void* d_ws, size_t ws_size,
                              hipStream_t stream) {
  (void)in_sizes; (void)n_in; (void)out_size; (void)ws_size;
  constexpr int Bb = 4, T = 2048, D = 1024, H = 16;
  constexpr int BT = Bb * T;  // 8192

  const float* x     = (const float*)d_in[0];
  const float* w_qkv = (const float*)d_in[1];
  const float* b_qkv = (const float*)d_in[2];
  const float* w_out = (const float*)d_in[3];
  const float* b_out = (const float*)d_in[4];
  float* out = (float*)d_out;

  // workspace layout (f16 elements)
  _Float16* qkv   = (_Float16*)d_ws;                   // BT*3D   (48 MB)
  _Float16* attn  = qkv   + (size_t)BT * 3 * D;        // BT*D    (16 MB)
  _Float16* xh    = attn  + (size_t)BT * D;            // BT*D    (16 MB)
  _Float16* wqkvT = xh    + (size_t)BT * D;            // 3D*D    ( 6 MB)
  _Float16* woutT = wqkvT + (size_t)3 * D * D;         // D*D     ( 2 MB)

  // prep: f16 copies / transposed weights
  cvt_f16_kernel<<<dim3((unsigned)((size_t)BT * D / 4 / 256)), dim3(256), 0, stream>>>(x, xh);
  transpose_cvt_kernel<<<dim3(3 * D / 64, D / 64), dim3(256), 0, stream>>>(w_qkv, wqkvT, D, 3 * D);
  transpose_cvt_kernel<<<dim3(D / 64, D / 64), dim3(256), 0, stream>>>(w_out, woutT, D, D);

  // QKV projection
  dim3 gQKV(3 * D / 128, BT / 64);
  gemm_wmma_kernel<_Float16>
      <<<gQKV, dim3(256), 0, stream>>>(xh, wqkvT, b_qkv, qkv, BT, 3 * D, D);

  // attention
  dim3 gAtt(T / 64, H, Bb);
  flash_attn_kernel<<<gAtt, dim3(128), 0, stream>>>(qkv, attn);

  // output projection
  dim3 gOut(D / 128, BT / 64);
  gemm_wmma_kernel<float>
      <<<gOut, dim3(256), 0, stream>>>(attn, woutT, b_out, out, BT, D, D);
}
